// GraphEncoder_1675037245478
// MI455X (gfx1250) — matrix operand order repaired and checked by reading, not compiled, not run on hardware
//
#include <hip/hip_runtime.h>
#include <hip/hip_bf16.h>

// ---------------------------------------------------------------------------
// GATv2 x2 for MI455X (gfx1250, wave32). Dense projections on WMMA f16->f32
// matrix cores; edge scatter phase with wave-per-edge vectorizable gathers and
// native f32 global atomics. Sizes: N=20000, E=320000, F=64, H=4, C=64, R=512.
// ---------------------------------------------------------------------------

typedef __attribute__((ext_vector_type(16))) _Float16 v16h;
typedef __attribute__((ext_vector_type(8)))  float    v8f;

#define NEG_SLOPE 0.2f

__device__ __forceinline__ float atom_add_f32(float* p, float v) {
    return __hip_atomic_fetch_add(p, v, __ATOMIC_RELAXED, __HIP_MEMORY_SCOPE_AGENT);
}

// Monotone float <-> uint mapping for atomicMax-based segment max.
__device__ __forceinline__ unsigned fkey(float f) {
    unsigned u = __float_as_uint(f);
    return u ^ ((unsigned)((int)u >> 31) | 0x80000000u);
}
__device__ __forceinline__ float fdecode(unsigned k) {
    unsigned u = k ^ ((unsigned)((int)(~k) >> 31) | 0x80000000u);
    return __uint_as_float(u);
}

// ---------------------------------------------------------------------------
// GEMM: Y[M,256] = X[M,64] * W[256,64]^T (+ bias). One wave per 16x16 output
// tile; K=64 done as two v_wmma_f32_16x16x32_f16 steps. M must be mult. of 16.
// A fragment (16-bit 16x32, ISA 7.12.2): lane l holds row M = l&15;
//   elem e -> K = (e>>3)*16 + (l>>4)*8 + (e&7).
// B fragment: lane l holds col N = l&15; elem e -> K = (e>>3)*16 + (l>>4)*8 + (e&7).
// C/D: lane l col N = l&15; vgpr r -> row M = r + 8*(l>>4).
// ---------------------------------------------------------------------------
__global__ void gemm_n256_wmma(const float* __restrict__ X,
                               const float* __restrict__ W,
                               const float* __restrict__ bias,
                               float* __restrict__ Y,
                               int M) {
    const int lane  = threadIdx.x & 31;
    const int hi    = lane >> 4;
    const int mrow  = blockIdx.x * 16 + (lane & 15);
    const int ncol  = blockIdx.y * 16 + (lane & 15);
    const float* xrow = X + (size_t)mrow * 64;
    const float* wrow = W + (size_t)ncol * 64;

    v8f c = {};
#pragma unroll
    for (int s = 0; s < 2; ++s) {
        v16h a, b;
#pragma unroll
        for (int e = 0; e < 16; ++e) {
            int ka = s * 32 + (e >> 3) * 16 + hi * 8 + (e & 7);
            int kb = s * 32 + (e >> 3) * 16 + hi * 8 + (e & 7);
            a[e] = (_Float16)xrow[ka];
            b[e] = (_Float16)wrow[kb];
        }
        c = __builtin_amdgcn_wmma_f32_16x16x32_f16(false, a, false, b,
                                                   (short)0, c, false, false);
    }
    const float bv = bias ? bias[ncol] : 0.0f;
#pragma unroll
    for (int r = 0; r < 8; ++r) {
        int m = blockIdx.x * 16 + r + 8 * hi;
        Y[(size_t)m * 256 + ncol] = c[r] + bv;
    }
}

// ---------------------------------------------------------------------------
// Degree + raw edge-attr scatter sum (64-wide). Wave per edge, 2 floats/lane.
// ---------------------------------------------------------------------------
__global__ void deg_sum_kernel(const int* __restrict__ dstA,
                               const int* __restrict__ relA,
                               const float* __restrict__ relations,
                               float* __restrict__ deg,
                               float* __restrict__ sumAttr,
                               int E) {
    int e = blockIdx.x * 8 + (threadIdx.x >> 5);
    if (e >= E) return;
    int lane = threadIdx.x & 31;
    int d = dstA[e], r = relA[e];
    if (lane == 0) atom_add_f32(&deg[d], 1.0f);
    const float* pr = relations + (size_t)r * 64 + lane * 2;
    float*       ps = sumAttr   + (size_t)d * 64 + lane * 2;
    atom_add_f32(&ps[0], pr[0]);
    atom_add_f32(&ps[1], pr[1]);
}

__global__ void mean_attr_kernel(const float* __restrict__ sumAttr,
                                 const float* __restrict__ deg,
                                 float* __restrict__ meanAttr, int total) {
    int i = blockIdx.x * 256 + threadIdx.x;
    if (i >= total) return;
    float dg = deg[i >> 6];
    dg = dg > 1.0f ? dg : 1.0f;
    meanAttr[i] = sumAttr[i] / dg;
}

// ---------------------------------------------------------------------------
// Per-edge attention logits. Wave per edge; lane covers 8 channels of one
// head (H*C = 256 = 32 lanes * 8). Butterfly-reduce within 8-lane head group.
// ---------------------------------------------------------------------------
__global__ void edge_alpha_kernel(const int* __restrict__ srcA,
                                  const int* __restrict__ dstA,
                                  const int* __restrict__ relA,
                                  const float* __restrict__ xl,
                                  const float* __restrict__ xr,
                                  const float* __restrict__ relEmb,
                                  const float* __restrict__ att,
                                  float* __restrict__ alpha,
                                  unsigned* __restrict__ amax,
                                  int E) {
    int e = blockIdx.x * 8 + (threadIdx.x >> 5);
    if (e >= E) return;
    int lane = threadIdx.x & 31;
    int s = srcA[e], d = dstA[e], r = relA[e];
    int base = lane * 8;
    int h = lane >> 3;
    const float* pl = xl     + (size_t)s * 256 + base;
    const float* pr = xr     + (size_t)d * 256 + base;
    const float* pe = relEmb + (size_t)r * 256 + base;
    const float* pa = att + base;  // att flat [H*C]; base already == h*64 + c0
    float partial = 0.0f;
#pragma unroll
    for (int j = 0; j < 8; ++j) {
        float m = pl[j] + pr[j] + pe[j];
        m = m > 0.0f ? m : NEG_SLOPE * m;
        partial += m * pa[j];
    }
    partial += __shfl_xor(partial, 1, 32);
    partial += __shfl_xor(partial, 2, 32);
    partial += __shfl_xor(partial, 4, 32);
    if ((lane & 7) == 0) {
        alpha[(size_t)e * 4 + h] = partial;
        atomicMax(&amax[(size_t)d * 4 + h], fkey(partial));
    }
}

// Self-loop logits: src=dst=n, edge embedding = meanEE[n]. Stored after the
// E real edges in the alpha buffer.
__global__ void loop_alpha_kernel(const float* __restrict__ xl,
                                  const float* __restrict__ xr,
                                  const float* __restrict__ meanEE,
                                  const float* __restrict__ att,
                                  float* __restrict__ alpha,
                                  unsigned* __restrict__ amax,
                                  int E, int Nn) {
    int n = blockIdx.x * 8 + (threadIdx.x >> 5);
    if (n >= Nn) return;
    int lane = threadIdx.x & 31;
    int base = lane * 8;
    int h = lane >> 3;
    const float* pl = xl     + (size_t)n * 256 + base;
    const float* pr = xr     + (size_t)n * 256 + base;
    const float* pe = meanEE + (size_t)n * 256 + base;
    const float* pa = att + base;
    float partial = 0.0f;
#pragma unroll
    for (int j = 0; j < 8; ++j) {
        float m = pl[j] + pr[j] + pe[j];
        m = m > 0.0f ? m : NEG_SLOPE * m;
        partial += m * pa[j];
    }
    partial += __shfl_xor(partial, 1, 32);
    partial += __shfl_xor(partial, 2, 32);
    partial += __shfl_xor(partial, 4, 32);
    if ((lane & 7) == 0) {
        alpha[(size_t)(E + n) * 4 + h] = partial;
        atomicMax(&amax[(size_t)n * 4 + h], fkey(partial));
    }
}

// exp(alpha - segmax) in place + segment-sum of denominators.
__global__ void exp_denom_kernel(const int* __restrict__ dstA,
                                 float* __restrict__ alpha,
                                 const unsigned* __restrict__ amax,
                                 float* __restrict__ denom,
                                 int E, int Nn) {
    int i = blockIdx.x * 256 + threadIdx.x;
    int tot = (E + Nn) * 4;
    if (i >= tot) return;
    int e2 = i >> 2, h = i & 3;
    int d = (e2 < E) ? dstA[e2] : (e2 - E);
    float mx = fdecode(amax[(size_t)d * 4 + h]);
    float ex = __expf(alpha[i] - mx);
    alpha[i] = ex;
    atom_add_f32(&denom[(size_t)d * 4 + h], ex);
}

// out[dst] += softmax_weight * xl[src]. Wave per edge, 8 channels/lane.
__global__ void scatter_out_kernel(const int* __restrict__ srcA,
                                   const int* __restrict__ dstA,
                                   const float* __restrict__ xl,
                                   const float* __restrict__ alpha,
                                   const float* __restrict__ denom,
                                   float* __restrict__ acc,
                                   int E, int Nn) {
    int e2 = blockIdx.x * 8 + (threadIdx.x >> 5);
    if (e2 >= E + Nn) return;
    int lane = threadIdx.x & 31;
    int s, d;
    if (e2 < E) { s = srcA[e2]; d = dstA[e2]; } else { s = d = e2 - E; }
    int h = lane >> 3;
    float w = alpha[(size_t)e2 * 4 + h] / denom[(size_t)d * 4 + h];
    int base = lane * 8;
    const float* px = xl  + (size_t)s * 256 + base;
    float*       pa = acc + (size_t)d * 256 + base;
#pragma unroll
    for (int j = 0; j < 8; ++j)
        atom_add_f32(&pa[j], px[j] * w);
}

// Head mean (concat=False) + bias.
__global__ void head_mean_kernel(const float* __restrict__ acc,
                                 const float* __restrict__ b,
                                 float* __restrict__ out, int total) {
    int i = blockIdx.x * 256 + threadIdx.x;
    if (i >= total) return;
    int n = i >> 6, c = i & 63;
    const float* p = acc + (size_t)n * 256 + c;
    out[i] = 0.25f * (p[0] + p[64] + p[128] + p[192]) + b[c];
}

// ---------------------------------------------------------------------------
extern "C" void kernel_launch(void* const* d_in, const int* in_sizes, int n_in,
                              void* d_out, int out_size, void* d_ws, size_t ws_size,
                              hipStream_t stream) {
    const float* x         = (const float*)d_in[0];
    const int*   ei        = (const int*)d_in[1];
    const float* relations = (const float*)d_in[2];
    const int*   relidx    = (const int*)d_in[3];

    const int E  = in_sizes[3];
    const int Nn = in_sizes[0] / 64;
    const int R  = in_sizes[2] / 64;
    const int* srcA = ei;
    const int* dstA = ei + E;

    // ---- carve workspace -------------------------------------------------
    char* wp = (char*)d_ws;
    auto carve = [&](size_t bytes) {
        char* p = wp;
        wp += (bytes + 255) & ~(size_t)255;
        return p;
    };
    float*    xl       = (float*)carve((size_t)Nn * 256 * 4);
    float*    xr       = (float*)carve((size_t)Nn * 256 * 4);
    float*    meanEE   = (float*)carve((size_t)Nn * 256 * 4);
    float*    acc      = (float*)carve((size_t)Nn * 256 * 4);
    float*    relEmb   = (float*)carve((size_t)R * 256 * 4);
    float*    meanAttr = (float*)carve((size_t)Nn * 64 * 4);
    float*    sumAttr  = (float*)carve((size_t)Nn * 64 * 4);
    float*    hbuf     = (float*)carve((size_t)Nn * 64 * 4);
    float*    deg      = (float*)carve((size_t)Nn * 4);
    unsigned* amax     = (unsigned*)carve((size_t)Nn * 4 * 4);
    float*    denom    = (float*)carve((size_t)Nn * 4 * 4);
    float*    alpha    = (float*)carve((size_t)(E + Nn) * 4 * 4);

    const int eBlocks  = (E + 7) / 8;       // wave-per-edge kernels
    const int nBlocks  = (Nn + 7) / 8;
    const int e2Blocks = (E + Nn + 7) / 8;
    const int ehBlocks = ((E + Nn) * 4 + 255) / 256;
    const int nfBlocks = (Nn * 64 + 255) / 256;

    // ---- edge-attr mean (shared by both layers) --------------------------
    hipMemsetAsync(deg, 0, (size_t)Nn * 4, stream);
    hipMemsetAsync(sumAttr, 0, (size_t)Nn * 64 * 4, stream);
    deg_sum_kernel<<<eBlocks, 256, 0, stream>>>(dstA, relidx, relations, deg, sumAttr, E);
    mean_attr_kernel<<<nfBlocks, 256, 0, stream>>>(sumAttr, deg, meanAttr, Nn * 64);

    // ---- two GATv2 layers ------------------------------------------------
    for (int l = 0; l < 2; ++l) {
        const float* Wl  = (const float*)d_in[4 + 7 * l];
        const float* bl  = (const float*)d_in[5 + 7 * l];
        const float* Wr  = (const float*)d_in[6 + 7 * l];
        const float* br  = (const float*)d_in[7 + 7 * l];
        const float* We  = (const float*)d_in[8 + 7 * l];
        const float* att = (const float*)d_in[9 + 7 * l];
        const float* bb  = (const float*)d_in[10 + 7 * l];
        const float* xin = (l == 0) ? x : hbuf;
        float* hout      = (l == 0) ? hbuf : (float*)d_out;

        dim3 gN(Nn / 16, 16), gR(R / 16, 16);
        gemm_n256_wmma<<<gN, 32, 0, stream>>>(xin, Wl, bl, xl, Nn);
        gemm_n256_wmma<<<gN, 32, 0, stream>>>(xin, Wr, br, xr, Nn);
        gemm_n256_wmma<<<gN, 32, 0, stream>>>(meanAttr, We, nullptr, meanEE, Nn);
        gemm_n256_wmma<<<gR, 32, 0, stream>>>(relations, We, nullptr, relEmb, R);

        hipMemsetAsync(amax, 0, (size_t)Nn * 4 * 4, stream);
        hipMemsetAsync(denom, 0, (size_t)Nn * 4 * 4, stream);
        hipMemsetAsync(acc, 0, (size_t)Nn * 256 * 4, stream);

        edge_alpha_kernel<<<eBlocks, 256, 0, stream>>>(srcA, dstA, relidx, xl, xr,
                                                       relEmb, att, alpha, amax, E);
        loop_alpha_kernel<<<nBlocks, 256, 0, stream>>>(xl, xr, meanEE, att,
                                                       alpha, amax, E, Nn);
        exp_denom_kernel<<<ehBlocks, 256, 0, stream>>>(dstA, alpha, amax, denom, E, Nn);
        scatter_out_kernel<<<e2Blocks, 256, 0, stream>>>(srcA, dstA, xl, alpha,
                                                         denom, acc, E, Nn);
        head_mean_kernel<<<nfBlocks, 256, 0, stream>>>(acc, bb, hout, Nn * 64);
    }

    // ---- tuple output: append relations unchanged ------------------------
    hipMemcpyAsync((float*)d_out + (size_t)Nn * 64, relations,
                   (size_t)R * 64 * 4, hipMemcpyDeviceToDevice, stream);
}
